// JointLocalProtoWeightingCE_12128987644149
// MI455X (gfx1250) — matrix operand order repaired and checked by reading, not compiled, not run on hardware
//
#include <hip/hip_runtime.h>

// ---------------------------------------------------------------------------
// JointLocalProtoWeightingCE on MI455X (gfx1250, wave32).
// Pipeline (all on `stream`, serialized by stream order):
//   1) k_softmax_ce : per-pixel softmax of inputs_plbl -> out_plbl [N,HW,C]
//                     and ce = -log(softmax(inputs)+eps)          [N,HW,C]
//   2) k_feat16     : feats [N,Ch,H,W] f32 -> feat16 [N,HW,Ch] f16 (WMMA feed)
//   3) k_spinit     : multi-hot flags, zero counters/seg_max/accums, proto_idx = HW
//   4) k_scatter    : atomicMax per (sp,class) on positive-probability bits;
//                     build per-superpixel valid-pixel lists (cap 64)
//   5) k_argmin     : atomicMin pixel index where score >= seg_max -> proto_idx
//   6) k_wmma_sim   : one wave per (n,s): 16-pixel x 32-class tile via
//                     v_wmma_f32_16x16x32_f16 over K=256, A loads software-
//                     pipelined one chunk ahead; then wave-wide scatter-softmax
//                     + weighted-CE accumulate
//   7) k_tail       : masked non-multi pixels use target one-hot weights; count
//   8) k_final      : out = cnt>0 ? loss/cnt : 0
// ---------------------------------------------------------------------------

namespace {

constexpr int   kN    = 2;
constexpr int   kC    = 19;
constexpr int   kHW   = 128 * 128;
constexpr int   kCh   = 256;
constexpr int   kS    = 2048;
constexpr int   kCap  = 64;
constexpr float kEps  = 1e-8f;
constexpr float kNeg  = -1e30f;
constexpr float kInvT = 10.0f;   // 1 / SIMW_TEMP

typedef __attribute__((ext_vector_type(16))) _Float16 v16h;
typedef __attribute__((ext_vector_type(8)))  _Float16 v8h;
typedef __attribute__((ext_vector_type(8)))  float    v8f;

// -------------------------------------------------- 1) dual softmax + CE
__global__ void k_softmax_ce(const float* __restrict__ in_plbl,
                             const float* __restrict__ in_main,
                             float* __restrict__ out_plbl,
                             float* __restrict__ ce) {
  int t = blockIdx.x * blockDim.x + threadIdx.x;
  if (t >= kN * kHW) return;
  int n = t / kHW, k = t - n * kHW;
  const float* a = in_plbl + (size_t)n * kC * kHW + k;
  const float* b = in_main + (size_t)n * kC * kHW + k;
  float va[kC], vb[kC];
  float ma = kNeg, mb = kNeg;
#pragma unroll
  for (int c = 0; c < kC; ++c) {
    va[c] = a[(size_t)c * kHW];
    vb[c] = b[(size_t)c * kHW];
    ma = fmaxf(ma, va[c]);
    mb = fmaxf(mb, vb[c]);
  }
  float sa = 0.f, sb = 0.f;
#pragma unroll
  for (int c = 0; c < kC; ++c) {
    va[c] = __expf(va[c] - ma); sa += va[c];
    vb[c] = __expf(vb[c] - mb); sb += vb[c];
  }
  float ra = 1.f / sa, rb = 1.f / sb;
  float* op = out_plbl + (size_t)t * kC;
  float* cp = ce + (size_t)t * kC;
#pragma unroll
  for (int c = 0; c < kC; ++c) {
    op[c] = va[c] * ra;
    cp[c] = -__logf(vb[c] * rb + kEps);
  }
}

// -------------------------------------------------- 2) feats -> f16, [N,HW,Ch]
__global__ void k_feat16(const float* __restrict__ f, _Float16* __restrict__ g) {
  size_t t = (size_t)blockIdx.x * blockDim.x + threadIdx.x;
  if (t >= (size_t)kN * kHW * kCh) return;
  int d = (int)(t % kCh);
  size_t nk = t / kCh;
  int k = (int)(nk % kHW);
  int n = (int)(nk / kHW);
  g[t] = (_Float16)f[((size_t)n * kCh + d) * kHW + k];   // L2 absorbs the stride
}

// -------------------------------------------------- 3) per-superpixel init
__global__ void k_spinit(const float* __restrict__ trg,
                         int* __restrict__ multi, int* __restrict__ counts,
                         unsigned* __restrict__ segmax, int* __restrict__ pidx,
                         float* __restrict__ lossAcc, int* __restrict__ cntAcc,
                         unsigned* __restrict__ anyv) {
  int t = blockIdx.x * blockDim.x + threadIdx.x;
  if (t >= kN * kS) return;
  const float* tr = trg + (size_t)t * kC;
  float s = 0.f;
#pragma unroll
  for (int c = 0; c < kC; ++c) s += tr[c];
  multi[t]  = (s > 1.f) ? 1 : 0;
  counts[t] = 0;
#pragma unroll
  for (int c = 0; c < kC; ++c) {
    segmax[(size_t)t * kC + c] = 0u;     // probs are positive; uint order == float order
    pidx[(size_t)t * kC + c]   = kHW;
  }
  if (t == 0) { *lossAcc = 0.f; *cntAcc = 0; anyv[0] = 0u; anyv[1] = 0u; }
}

// -------------------------------------------------- 4) segment max + pixel lists
__global__ void k_scatter(const int* __restrict__ sp,
                          const unsigned char* __restrict__ mask,
                          const int* __restrict__ multi,
                          const float* __restrict__ out_plbl,
                          unsigned* __restrict__ segmax,
                          int* __restrict__ counts,
                          int* __restrict__ plist,
                          unsigned* __restrict__ anyv) {
  int t = blockIdx.x * blockDim.x + threadIdx.x;
  if (t >= kN * kHW) return;
  int n = t / kHW, k = t - n * kHW;
  int s = sp[t];
  int bs = n * kS + s;
  if (!mask[t] || !multi[bs]) return;
  anyv[n] = 1u;                                   // racy same-value store, fine
  const float* op = out_plbl + (size_t)t * kC;
  unsigned* sm = segmax + (size_t)bs * kC;
#pragma unroll
  for (int c = 0; c < kC; ++c)
    atomicMax(&sm[c], __float_as_uint(op[c]));
  int pos = atomicAdd(&counts[bs], 1);
  if (pos < kCap) plist[(size_t)bs * kCap + pos] = k;
}

// -------------------------------------------------- 5) segment argmin (proto idx)
__global__ void k_argmin(const int* __restrict__ sp,
                         const unsigned char* __restrict__ mask,
                         const int* __restrict__ multi,
                         const float* __restrict__ out_plbl,
                         const unsigned* __restrict__ segmax,
                         int* __restrict__ pidx) {
  int t = blockIdx.x * blockDim.x + threadIdx.x;
  if (t >= kN * kHW) return;
  int n = t / kHW, k = t - n * kHW;
  int s = sp[t];
  int bs = n * kS + s;
  if (!mask[t] || !multi[bs]) return;
  const float* op = out_plbl + (size_t)t * kC;
  const unsigned* sm = segmax + (size_t)bs * kC;
#pragma unroll
  for (int c = 0; c < kC; ++c)
    if (op[c] >= __uint_as_float(sm[c]))
      atomicMin(&pidx[(size_t)bs * kC + c], k);
}

// -------------------------------------------------- 6) WMMA sim + scatter-softmax
// One wave per (n,s). A = 16 valid pixels' features (16xK f16), B = prototype
// features (Kx16 f16, columns = classes; two halves for classes 0..15 / 16..31).
// VGPR fragment layouts (C/D per CDNA5 ISA 7.12.2; A/B k order is this code's own):
//   A 16x32 f16: lanes<16 -> M=lane, K 0..15; lanes>=16 -> K 16..31 (contiguous, not ISA table order)
//   B 32x16 f16: lane n -> column n, lanes<16 K 0..15, lanes>=16 K 16..31 (contiguous, as A)
//   C/D 16x16 f32: vgpr r: lanes<16 -> M=r, N=lane; lanes>=16 -> M=8+r, N=lane-16
__global__ __launch_bounds__(32)
void k_wmma_sim(const _Float16* __restrict__ feat16,
                const int* __restrict__ counts,
                const int* __restrict__ plist,
                const int* __restrict__ pidx,
                const float* __restrict__ trg,
                const float* __restrict__ ce,
                float* __restrict__ lossAcc) {
  int b = blockIdx.x;                       // n*kS + s
  int n = b / kS;
  int cnt = counts[b];                      // wave-uniform
  if (cnt <= 0) return;
  if (cnt > kCap) cnt = kCap;
  int L    = threadIdx.x;
  int half = L >> 4;
  int lc   = L & 15;

  const _Float16* fb = feat16 + (size_t)n * kHW * kCh;
  const int* prow = pidx + (size_t)b * kC;
  int c0 = lc, c1 = 16 + lc;
  int p0 = prow[c0];             p0 = p0 < 0 ? 0 : (p0 >= kHW ? kHW - 1 : p0);
  int p1 = (c1 < kC) ? prow[c1] : 0;
  p1 = p1 < 0 ? 0 : (p1 >= kHW ? kHW - 1 : p1);
  const _Float16* b0p = fb + (size_t)p0 * kCh;
  const _Float16* b1p = fb + (size_t)p1 * kCh;   // cols >= kC: garbage, ignored

  const float* trow = trg + (size_t)b * kC;
  bool act0 = trow[c0] > 0.f;
  bool act1 = (c1 < kC) && (trow[c1] > 0.f);

  int offA = half ? 16 : 0;
  int offB = half ? 16 : 0;
  const int* pl = plist + (size_t)b * kCap;
  float lsum = 0.f;

  int ntile = (cnt + 15) >> 4;
  for (int t = 0; t < ntile; ++t) {
    int m  = (t << 4) + lc;
    int mi = m < cnt ? m : cnt - 1;           // pad rows duplicate last pixel
    int pix = pl[mi];
    const _Float16* arow = fb + (size_t)pix * kCh;

    // Software-pipelined A-fragment stream: loads run one K-chunk ahead of the
    // WMMA pair consuming them, hiding L2 latency behind the XDL pipeline.
    v8h alo = *(const v8h*)(arow + offA);
    v8h ahi = *(const v8h*)(arow + 8 + offA);
    v8f acc0 = {}; v8f acc1 = {};
#pragma unroll
    for (int k0 = 0; k0 < kCh; k0 += 32) {
      v8h nlo = alo, nhi = ahi;
      if (k0 + 32 < kCh) {
        nlo = *(const v8h*)(arow + k0 + 32 + offA);
        nhi = *(const v8h*)(arow + k0 + 40 + offA);
      }
      v16h A  = __builtin_shufflevector(alo, ahi,
                 0,1,2,3,4,5,6,7,8,9,10,11,12,13,14,15);
      v16h B0 = *(const v16h*)(b0p + k0 + offB);
      v16h B1 = *(const v16h*)(b1p + k0 + offB);
      acc0 = __builtin_amdgcn_wmma_f32_16x16x32_f16(false, A, false, B0,
                                                    (short)0, acc0, false, false);
      acc1 = __builtin_amdgcn_wmma_f32_16x16x32_f16(false, A, false, B1,
                                                    (short)0, acc1, false, false);
      alo = nlo; ahi = nhi;
    }

    int rows = cnt - (t << 4); if (rows > 16) rows = 16;
#pragma unroll
    for (int mm = 0; mm < 16; ++mm) {
      if (mm >= rows) break;                  // uniform
      bool ing = (mm < 8) ? (half == 0) : (half == 1);
      float x0 = (ing && act0) ? acc0[mm & 7] * kInvT : kNeg;
      float x1 = (ing && act1) ? acc1[mm & 7] * kInvT : kNeg;
      float mx = fmaxf(x0, x1);
      for (int off = 16; off > 0; off >>= 1)
        mx = fmaxf(mx, __shfl_xor(mx, off));
      float e0 = (ing && act0) ? __expf(x0 - mx) : 0.f;
      float e1 = (ing && act1) ? __expf(x1 - mx) : 0.f;
      float es = e0 + e1;
      for (int off = 16; off > 0; off >>= 1)
        es += __shfl_xor(es, off);
      float inv = 1.f / es;                   // >=2 active classes guaranteed
      int pixm = pl[(t << 4) + mm];
      const float* crow = ce + ((size_t)n * kHW + pixm) * kC;
      float contrib = 0.f;
      if (ing && act0) contrib += e0 * inv * crow[c0];
      if (ing && act1) contrib += e1 * inv * crow[c1];
      for (int off = 16; off > 0; off >>= 1)
        contrib += __shfl_xor(contrib, off);
      if (L == 0) lsum += contrib;
    }
  }
  if (L == 0) atomicAdd(lossAcc, lsum);
}

// -------------------------------------------------- 7) one-hot pixels + count
__global__ void k_tail(const int* __restrict__ sp,
                       const unsigned char* __restrict__ mask,
                       const int* __restrict__ multi,
                       const float* __restrict__ trg,
                       const float* __restrict__ ce,
                       const unsigned* __restrict__ anyv,
                       float* __restrict__ lossAcc,
                       int* __restrict__ cntAcc) {
  int t = blockIdx.x * blockDim.x + threadIdx.x;
  float contrib = 0.f;
  int cn = 0;
  if (t < kN * kHW) {
    int n = t / kHW;
    if (anyv[n] && mask[t]) {
      cn = 1;
      int s = sp[t];
      if (!multi[n * kS + s]) {               // invalid pixel: w = target row
        const float* tr = trg + ((size_t)n * kS + s) * kC;
        const float* cr = ce + (size_t)t * kC;
#pragma unroll
        for (int c = 0; c < kC; ++c) contrib += tr[c] * cr[c];
      }
    }
  }
  for (int off = 16; off > 0; off >>= 1) {
    contrib += __shfl_xor(contrib, off);
    cn      += __shfl_xor(cn, off);
  }
  if ((threadIdx.x & 31) == 0) {
    if (contrib != 0.f) atomicAdd(lossAcc, contrib);
    if (cn)             atomicAdd(cntAcc, cn);
  }
}

// -------------------------------------------------- 8) final scalar
__global__ void k_final(const float* __restrict__ lossAcc,
                        const int* __restrict__ cntAcc,
                        float* __restrict__ out) {
  if (threadIdx.x == 0 && blockIdx.x == 0) {
    int c = *cntAcc;
    out[0] = (c > 0) ? (*lossAcc / (float)c) : 0.f;
  }
}

} // namespace

extern "C" void kernel_launch(void* const* d_in, const int* /*in_sizes*/, int /*n_in*/,
                              void* d_out, int /*out_size*/, void* d_ws, size_t /*ws_size*/,
                              hipStream_t stream) {
  const float*         in_plbl = (const float*)d_in[0];
  const float*         feats   = (const float*)d_in[1];
  const float*         in_main = (const float*)d_in[2];
  const float*         trg     = (const float*)d_in[3];
  const int*           sp      = (const int*)d_in[4];
  const unsigned char* mask    = (const unsigned char*)d_in[5];   // jax bool = 1 byte
  float* out = (float*)d_out;

  char* ws = (char*)d_ws;
  size_t off = 0;
  float*    out_plbl = (float*)(ws + off);    off += (size_t)kN * kHW * kC * 4;
  float*    ce       = (float*)(ws + off);    off += (size_t)kN * kHW * kC * 4;
  _Float16* f16      = (_Float16*)(ws + off); off += (size_t)kN * kHW * kCh * 2;
  unsigned* segmax   = (unsigned*)(ws + off); off += (size_t)kN * kS * kC * 4;
  int*      pidx     = (int*)(ws + off);      off += (size_t)kN * kS * kC * 4;
  int*      counts   = (int*)(ws + off);      off += (size_t)kN * kS * 4;
  int*      multi    = (int*)(ws + off);      off += (size_t)kN * kS * 4;
  int*      plist    = (int*)(ws + off);      off += (size_t)kN * kS * kCap * 4;
  float*    lossAcc  = (float*)(ws + off);
  int*      cntAcc   = (int*)(ws + off + 4);
  unsigned* anyv     = (unsigned*)(ws + off + 8);

  const int thr = 256;
  const int pixBlocks = (kN * kHW + thr - 1) / thr;
  const size_t fElems = (size_t)kN * kHW * kCh;

  hipLaunchKernelGGL(k_softmax_ce, dim3(pixBlocks), dim3(thr), 0, stream,
                     in_plbl, in_main, out_plbl, ce);
  hipLaunchKernelGGL(k_feat16, dim3((unsigned)((fElems + thr - 1) / thr)), dim3(thr), 0, stream,
                     feats, f16);
  hipLaunchKernelGGL(k_spinit, dim3((kN * kS + thr - 1) / thr), dim3(thr), 0, stream,
                     trg, multi, counts, segmax, pidx, lossAcc, cntAcc, anyv);
  hipLaunchKernelGGL(k_scatter, dim3(pixBlocks), dim3(thr), 0, stream,
                     sp, mask, multi, out_plbl, segmax, counts, plist, anyv);
  hipLaunchKernelGGL(k_argmin, dim3(pixBlocks), dim3(thr), 0, stream,
                     sp, mask, multi, out_plbl, segmax, pidx);
  hipLaunchKernelGGL(k_wmma_sim, dim3(kN * kS), dim3(32), 0, stream,
                     f16, counts, plist, pidx, trg, ce, lossAcc);
  hipLaunchKernelGGL(k_tail, dim3(pixBlocks), dim3(thr), 0, stream,
                     sp, mask, multi, trg, ce, anyv, lossAcc, cntAcc);
  hipLaunchKernelGGL(k_final, dim3(1), dim3(1), 0, stream, lossAcc, cntAcc, out);
}
